// TransformerEncoder_21921513079125
// MI455X (gfx1250) — compile-verified, matrix operand order checked by reading, not run on hardware
//
#include <hip/hip_runtime.h>
#include <math.h>
#include <stdint.h>

typedef __bf16 bf16;
typedef __attribute__((ext_vector_type(16))) bf16  v16bf;
typedef __attribute__((ext_vector_type(8)))  bf16  v8bf;
typedef __attribute__((ext_vector_type(8)))  float v8f;

#define B_    8
#define N_    1024
#define F_    512
#define H_    8
#define DH_   64
#define MLPH_ 2048
#define TOK_  (B_ * N_)

// ---------------------------------------------------------------------------
// CDNA5 async global->LDS copy (tracked by ASYNCcnt, bypasses VGPRs)
// ---------------------------------------------------------------------------
__device__ __forceinline__ uint32_t lds_off(const void* p) {
  // Generic pointers to LDS carry the LDS byte offset in bits [31:0].
  return (uint32_t)(size_t)p;
}
__device__ __forceinline__ void async_cp16(uint32_t ldsoff, const void* gptr) {
  asm volatile("global_load_async_to_lds_b128 %0, %1, off"
               :: "v"(ldsoff), "v"(gptr)
               : "memory");
}
__device__ __forceinline__ void wait_async0() {
  asm volatile("s_wait_asynccnt 0" ::: "memory");
}
__device__ __forceinline__ void wait_async6() {
  asm volatile("s_wait_asynccnt 6" ::: "memory");
}

// ---------------------------------------------------------------------------
// WMMA helpers (CDNA5: D = A(16x32 bf16) * B(32x16 bf16) + C(16x16 f32))
// ---------------------------------------------------------------------------
__device__ __forceinline__ v8f wmma_bf16(v16bf a, v16bf b, v8f c) {
  return __builtin_amdgcn_wmma_f32_16x16x32_bf16(false, a, false, b, (short)0, c,
                                                 false, false);
}

// 16-bit A/B fragment per ISA layout: lanes 0-15 take K = {0..7,16..23}+base,
// lanes 16-31 take K = {8..15,24..31}+base; kb = (lane>=16)?8:0.
__device__ __forceinline__ v16bf load_frag(const bf16* rowptr, int kb) {
  union { v16bf v; v8bf h[2]; } u;
  u.h[0] = *reinterpret_cast<const v8bf*>(rowptr + kb);
  u.h[1] = *reinterpret_cast<const v8bf*>(rowptr + kb + 16);
  return u.v;
}

__device__ __forceinline__ float gelu_exact(float x) {
  return 0.5f * x * (1.0f + erff(x * 0.70710678118654752440f));
}

// ---------------------------------------------------------------------------
// fp32 [K][N] -> bf16 [N][K] transpose (weights), 32x32 tiles via LDS
// ---------------------------------------------------------------------------
__global__ __launch_bounds__(256)
void cvt_transpose_bf16(const float* __restrict__ in, bf16* __restrict__ out,
                        int K, int N) {
  __shared__ float t[32][33];
  const int k0 = blockIdx.x * 32, n0 = blockIdx.y * 32;
  const int tx = threadIdx.x & 31, ty = threadIdx.x >> 5;  // ty = 0..7
#pragma unroll
  for (int i = ty; i < 32; i += 8) t[i][tx] = in[(size_t)(k0 + i) * N + n0 + tx];
  __syncthreads();
#pragma unroll
  for (int i = ty; i < 32; i += 8)
    out[(size_t)(n0 + i) * K + k0 + tx] = (bf16)t[tx][i];
}

// ---------------------------------------------------------------------------
// LayerNorm over F_=512, emits bf16 activations. One block (256 thr) per row.
// ---------------------------------------------------------------------------
__global__ __launch_bounds__(256)
void layernorm_to_bf16(const float* __restrict__ x, const float* __restrict__ g,
                       const float* __restrict__ beta, bf16* __restrict__ out) {
  const int row = blockIdx.x;
  const int tid = threadIdx.x;
  const float* xr = x + (size_t)row * F_;
  float v0 = xr[tid], v1 = xr[tid + 256];
  float s  = v0 + v1;
  float sq = v0 * v0 + v1 * v1;
#pragma unroll
  for (int off = 1; off < 32; off <<= 1) {
    s  += __shfl_xor(s, off, 32);
    sq += __shfl_xor(sq, off, 32);
  }
  __shared__ float red[2][8];
  if ((tid & 31) == 0) { red[0][tid >> 5] = s; red[1][tid >> 5] = sq; }
  __syncthreads();
  float ts = 0.f, tq = 0.f;
#pragma unroll
  for (int i = 0; i < 8; ++i) { ts += red[0][i]; tq += red[1][i]; }
  const float mean = ts * (1.0f / (float)F_);
  const float var  = tq * (1.0f / (float)F_) - mean * mean;
  const float rstd = rsqrtf(var + 1e-5f);
  bf16* orow = out + (size_t)row * F_;
  orow[tid]       = (bf16)((v0 - mean) * rstd * g[tid]       + beta[tid]);
  orow[tid + 256] = (bf16)((v1 - mean) * rstd * g[tid + 256] + beta[tid + 256]);
}

// ---------------------------------------------------------------------------
// bf16 WMMA GEMM: C = A[MxK] * WT[NxK]^T + bias (+gelu) (+resid)
// Block tile 128x64, 8 waves (4x2), wave tile 32x32, K step 64.
// Async global->LDS staging, double-buffered. Output fp32 and/or bf16.
// ---------------------------------------------------------------------------
__global__ __launch_bounds__(256)
void gemm_bf16(const bf16* __restrict__ A, const bf16* __restrict__ WT,
               const float* __restrict__ bias, const float* __restrict__ resid,
               float* __restrict__ outF, bf16* __restrict__ outB,
               int M, int K, int N, int do_gelu) {
  __shared__ __attribute__((aligned(16))) bf16 sA[2][128][72];
  __shared__ __attribute__((aligned(16))) bf16 sBT[2][64][72];
  const int m0   = blockIdx.x * 128;
  const int n0   = blockIdx.y * 64;
  const int tid  = threadIdx.x;
  const int wave = tid >> 5;
  const int lane = tid & 31;
  const int half = lane >> 4;
  const int ln   = lane & 15;
  const int wm   = wave >> 1;  // 0..3
  const int wn   = wave & 1;   // 0..1
  const int kb   = half * 8;

  // Per-thread async-copy assignments (16B chunks):
  // A tile 128x64 = 1024 chunks -> 4/thread; B tile 64x64 = 512 -> 2/thread.
  const int arow = tid >> 1, acb = (tid & 1) * 4;
  const int brow = tid >> 2, bcb = (tid & 3) * 2;
  auto issue_tile = [&](int buf, int k0) {
#pragma unroll
    for (int i = 0; i < 4; ++i)
      async_cp16(lds_off(&sA[buf][arow][(acb + i) * 8]),
                 A + (size_t)(m0 + arow) * K + k0 + (acb + i) * 8);
#pragma unroll
    for (int i = 0; i < 2; ++i)
      async_cp16(lds_off(&sBT[buf][brow][(bcb + i) * 8]),
                 WT + (size_t)(n0 + brow) * K + k0 + (bcb + i) * 8);
  };

  v8f c[2][2];
#pragma unroll
  for (int i = 0; i < 2; ++i)
#pragma unroll
    for (int j = 0; j < 2; ++j)
      c[i][j] = (v8f){0.f, 0.f, 0.f, 0.f, 0.f, 0.f, 0.f, 0.f};

  issue_tile(0, 0);
  const int nsteps = K >> 6;
  for (int s = 0; s < nsteps; ++s) {
    if (s + 1 < nsteps) {
      issue_tile((s + 1) & 1, (s + 1) * 64);
      wait_async6();  // oldest 6 (current tile) complete; next 6 in flight
    } else {
      wait_async0();
    }
    __syncthreads();
    const int cur = s & 1;
    v16bf a[2][2], bfr[2][2];
#pragma unroll
    for (int i = 0; i < 2; ++i) {
      const bf16* rp = &sA[cur][wm * 32 + i * 16 + ln][0];
      a[i][0] = load_frag(rp, kb);
      a[i][1] = load_frag(rp + 32, kb);
    }
#pragma unroll
    for (int j = 0; j < 2; ++j) {
      const bf16* rp = &sBT[cur][wn * 32 + j * 16 + ln][0];
      bfr[j][0] = load_frag(rp, kb);
      bfr[j][1] = load_frag(rp + 32, kb);
    }
#pragma unroll
    for (int i = 0; i < 2; ++i)
#pragma unroll
      for (int j = 0; j < 2; ++j) {
        c[i][j] = wmma_bf16(a[i][0], bfr[j][0], c[i][j]);
        c[i][j] = wmma_bf16(a[i][1], bfr[j][1], c[i][j]);
      }
    __syncthreads();  // all waves done with buf `cur` before it is re-filled
  }

  // Epilogue. C layout: VGPR r -> row r (lanes 0-15) / row r+8 (lanes 16-31).
#pragma unroll
  for (int i = 0; i < 2; ++i) {
#pragma unroll
    for (int j = 0; j < 2; ++j) {
#pragma unroll
      for (int r = 0; r < 8; ++r) {
        int row = m0 + wm * 32 + i * 16 + r + half * 8;
        int col = n0 + wn * 32 + j * 16 + ln;
        float v = c[i][j][r] + bias[col];
        if (do_gelu) v = gelu_exact(v);
        size_t idx = (size_t)row * N + col;
        if (resid) v += resid[idx];
        if (outF) outF[idx] = v;
        if (outB) outB[idx] = (bf16)v;
      }
    }
  }
}

// ---------------------------------------------------------------------------
// Flash attention: grid (B*H, N/64), 128 threads (4 waves).
// Each block: 64 query rows; loop over 16 K/V tiles of 64 keys.
// q/k/v/attn are bf16 [B*N][F] row-major; head h occupies cols h*64..h*64+63.
// ---------------------------------------------------------------------------
__global__ __launch_bounds__(128)
void flash_attn(const bf16* __restrict__ q, const bf16* __restrict__ k,
                const bf16* __restrict__ v, bf16* __restrict__ attn) {
  __shared__ __attribute__((aligned(16))) bf16 sQ[64][72];
  __shared__ __attribute__((aligned(16))) bf16 sK[64][72];
  __shared__ __attribute__((aligned(16))) bf16 sVT[64][72];  // sVT[d][key]
  __shared__ __attribute__((aligned(16))) bf16 sP[64][72];
  const int bh   = blockIdx.x;
  const int b    = bh / H_;
  const int h    = bh % H_;
  const int q0   = blockIdx.y * 64;
  const int tid  = threadIdx.x;
  const int wave = tid >> 5;
  const int lane = tid & 31;
  const int half = lane >> 4;
  const int ln   = lane & 15;
  const int kb   = half * 8;
  const float scale = 0.044194173824159220f;  // 1/sqrt(F_) per reference

  const int r_ld = tid >> 1, hc_ld = (tid & 1) * 32;  // 128 thr: 4 chunks each
  {  // async-load Q tile (64x64 bf16)
    const bf16* src = q + (size_t)(b * N_ + q0 + r_ld) * F_ + h * DH_ + hc_ld;
#pragma unroll
    for (int i = 0; i < 4; ++i)
      async_cp16(lds_off(&sQ[r_ld][hc_ld + i * 8]), src + i * 8);
  }

  v8f o[4];
  float m_run[8], l_run[8];
#pragma unroll
  for (int j = 0; j < 4; ++j) o[j] = (v8f){0.f, 0.f, 0.f, 0.f, 0.f, 0.f, 0.f, 0.f};
#pragma unroll
  for (int r = 0; r < 8; ++r) { m_run[r] = -1e30f; l_run[r] = 0.f; }

  for (int kc = 0; kc < N_; kc += 64) {
    __syncthreads();  // prior iteration done reading sK/sVT/sP
    {  // async-load K tile
      const bf16* src = k + (size_t)(b * N_ + kc + r_ld) * F_ + h * DH_ + hc_ld;
#pragma unroll
      for (int i = 0; i < 4; ++i)
        async_cp16(lds_off(&sK[r_ld][hc_ld + i * 8]), src + i * 8);
    }
    {  // V tile, transposed into sVT[d][key] (scalar; transpose in-flight)
      const bf16* src = v + (size_t)(b * N_ + kc + r_ld) * F_ + h * DH_ + hc_ld;
#pragma unroll
      for (int d = 0; d < 32; ++d) sVT[hc_ld + d][r_ld] = src[d];
    }
    wait_async0();  // covers K tile (and Q tile on first iteration)
    __syncthreads();

    // S strip (rows wave*16..+15 x 64 keys) = Q K^T, K-dim = DH_ = 2x32
    v16bf a0 = load_frag(&sQ[wave * 16 + ln][0], kb);
    v16bf a1 = load_frag(&sQ[wave * 16 + ln][32], kb);
    v8f s[4];
#pragma unroll
    for (int j = 0; j < 4; ++j) {
      v16bf b0 = load_frag(&sK[j * 16 + ln][0], kb);
      v16bf b1 = load_frag(&sK[j * 16 + ln][32], kb);
      v8f acc = (v8f){0.f, 0.f, 0.f, 0.f, 0.f, 0.f, 0.f, 0.f};
      acc = wmma_bf16(a0, b0, acc);
      acc = wmma_bf16(a1, b1, acc);
      s[j] = acc;
    }

    // Online softmax: row (r, lane-half) spread across 16 lanes of that half
#pragma unroll
    for (int r = 0; r < 8; ++r) {
      float mx = -1e30f;
#pragma unroll
      for (int j = 0; j < 4; ++j) mx = fmaxf(mx, s[j][r] * scale);
#pragma unroll
      for (int off = 1; off < 16; off <<= 1) mx = fmaxf(mx, __shfl_xor(mx, off, 16));
      float mnew = fmaxf(m_run[r], mx);
      float corr = __expf(m_run[r] - mnew);
      l_run[r] *= corr;
#pragma unroll
      for (int j = 0; j < 4; ++j) o[j][r] *= corr;
      float rs = 0.f;
#pragma unroll
      for (int j = 0; j < 4; ++j) {
        float pij = __expf(s[j][r] * scale - mnew);
        s[j][r] = pij;
        rs += pij;
      }
#pragma unroll
      for (int off = 1; off < 16; off <<= 1) rs += __shfl_xor(rs, off, 16);
      l_run[r] += rs;
      m_run[r] = mnew;
    }

    // Stage P (bf16) through LDS to reshape C-layout -> A-fragment layout
#pragma unroll
    for (int j = 0; j < 4; ++j)
#pragma unroll
      for (int r = 0; r < 8; ++r)
        sP[wave * 16 + r + half * 8][j * 16 + ln] = (bf16)s[j][r];
    __syncthreads();

    // O += P * V  (K-dim = 64 keys = 2x32)
    v16bf p0 = load_frag(&sP[wave * 16 + ln][0], kb);
    v16bf p1 = load_frag(&sP[wave * 16 + ln][32], kb);
#pragma unroll
    for (int j = 0; j < 4; ++j) {
      v16bf vb0 = load_frag(&sVT[j * 16 + ln][0], kb);
      v16bf vb1 = load_frag(&sVT[j * 16 + ln][32], kb);
      o[j] = wmma_bf16(p0, vb0, o[j]);
      o[j] = wmma_bf16(p1, vb1, o[j]);
    }
  }

  // Normalize by l and write bf16 attn output
#pragma unroll
  for (int j = 0; j < 4; ++j)
#pragma unroll
    for (int r = 0; r < 8; ++r) {
      int row = q0 + wave * 16 + r + half * 8;
      int d   = j * 16 + ln;
      attn[(size_t)(b * N_ + row) * F_ + h * DH_ + d] = (bf16)(o[j][r] / l_run[r]);
    }
}

// ---------------------------------------------------------------------------
// Host launch
// ---------------------------------------------------------------------------
extern "C" void kernel_launch(void* const* d_in, const int* in_sizes, int n_in,
                              void* d_out, int out_size, void* d_ws, size_t ws_size,
                              hipStream_t stream) {
  (void)in_sizes; (void)n_in; (void)out_size; (void)ws_size;
  const float* x   = (const float*)d_in[0];
  const float* wq  = (const float*)d_in[1];
  const float* bq  = (const float*)d_in[2];
  const float* wk  = (const float*)d_in[3];
  const float* bk  = (const float*)d_in[4];
  const float* wv  = (const float*)d_in[5];
  const float* bv  = (const float*)d_in[6];
  const float* wo  = (const float*)d_in[7];
  const float* bo  = (const float*)d_in[8];
  const float* g1  = (const float*)d_in[9];
  const float* be1 = (const float*)d_in[10];
  const float* g2  = (const float*)d_in[11];
  const float* be2 = (const float*)d_in[12];
  const float* w1  = (const float*)d_in[13];
  const float* b1  = (const float*)d_in[14];
  const float* w2  = (const float*)d_in[15];
  const float* b2  = (const float*)d_in[16];
  float* out = (float*)d_out;

  char* p = (char*)d_ws;
  auto carve = [&](size_t bytes) {
    void* r = (void*)p;
    p += (bytes + 255) & ~(size_t)255;
    return r;
  };
  bf16*  h1   = (bf16*)carve((size_t)TOK_ * F_ * 2);
  bf16*  qb   = (bf16*)carve((size_t)TOK_ * F_ * 2);
  bf16*  kbuf = (bf16*)carve((size_t)TOK_ * F_ * 2);
  bf16*  vb   = (bf16*)carve((size_t)TOK_ * F_ * 2);
  bf16*  attn = (bf16*)carve((size_t)TOK_ * F_ * 2);
  float* res1 = (float*)carve((size_t)TOK_ * F_ * 4);
  bf16*  h2   = (bf16*)carve((size_t)TOK_ * F_ * 2);
  bf16*  hid  = (bf16*)carve((size_t)TOK_ * MLPH_ * 2);
  bf16*  wqT  = (bf16*)carve((size_t)F_ * F_ * 2);      // [N][K] transposed
  bf16*  wkT  = (bf16*)carve((size_t)F_ * F_ * 2);
  bf16*  wvT  = (bf16*)carve((size_t)F_ * F_ * 2);
  bf16*  woT  = (bf16*)carve((size_t)F_ * F_ * 2);
  bf16*  w1T  = (bf16*)carve((size_t)F_ * MLPH_ * 2);   // [2048][512]
  bf16*  w2T  = (bf16*)carve((size_t)MLPH_ * F_ * 2);   // [512][2048]

  dim3 tFF(F_ / 32, F_ / 32);
  cvt_transpose_bf16<<<tFF, 256, 0, stream>>>(wq, wqT, F_, F_);
  cvt_transpose_bf16<<<tFF, 256, 0, stream>>>(wk, wkT, F_, F_);
  cvt_transpose_bf16<<<tFF, 256, 0, stream>>>(wv, wvT, F_, F_);
  cvt_transpose_bf16<<<tFF, 256, 0, stream>>>(wo, woT, F_, F_);
  cvt_transpose_bf16<<<dim3(F_ / 32, MLPH_ / 32), 256, 0, stream>>>(w1, w1T, F_, MLPH_);
  cvt_transpose_bf16<<<dim3(MLPH_ / 32, F_ / 32), 256, 0, stream>>>(w2, w2T, MLPH_, F_);

  layernorm_to_bf16<<<TOK_, 256, 0, stream>>>(x, g1, be1, h1);

  dim3 gFF(TOK_ / 128, F_ / 64);
  gemm_bf16<<<gFF, 256, 0, stream>>>(h1, wqT, bq, nullptr, nullptr, qb, TOK_, F_, F_, 0);
  gemm_bf16<<<gFF, 256, 0, stream>>>(h1, wkT, bk, nullptr, nullptr, kbuf, TOK_, F_, F_, 0);
  gemm_bf16<<<gFF, 256, 0, stream>>>(h1, wvT, bv, nullptr, nullptr, vb, TOK_, F_, F_, 0);

  flash_attn<<<dim3(B_ * H_, N_ / 64), 128, 0, stream>>>(qb, kbuf, vb, attn);

  // o = attn @ wo + bo, + residual x  -> res1 (fp32)
  gemm_bf16<<<gFF, 256, 0, stream>>>(attn, woT, bo, x, res1, nullptr, TOK_, F_, F_, 0);

  layernorm_to_bf16<<<TOK_, 256, 0, stream>>>(res1, g2, be2, h2);

  // hidden = gelu(h2 @ w1 + b1) -> bf16
  gemm_bf16<<<dim3(TOK_ / 128, MLPH_ / 64), 256, 0, stream>>>(
      h2, w1T, b1, nullptr, nullptr, hid, TOK_, F_, MLPH_, 1);
  // out = gelu(hidden @ w2 + b2) + res1 -> fp32 d_out
  gemm_bf16<<<gFF, 256, 0, stream>>>(hid, w2T, b2, res1, out, nullptr, TOK_, MLPH_, F_, 1);
}
